// KANLayer_30717606101109
// MI455X (gfx1250) — matrix operand order repaired and checked by reading, not compiled, not run on hardware
//
#include <hip/hip_runtime.h>

// ---------------------------------------------------------------------------
// KAN layer as one fused bf16 WMMA GEMM on gfx1250 (wave32, WGP).
//   out[b,o] = silu(x) @ Wb  +  basis(x) @ Ws,   basis built on-the-fly.
// Spline K ordered k = g*1024 + i so each BK=32 chunk has one mu[g].
// Double-buffered LDS: one barrier per K-step; next tile's VALU
// (basis/silu + bf16 convert) overlaps current tile's WMMAs.
// Raw v_exp_f32 (no denorm fixup): flush-to-zero is correct here since the
// result is truncated to bf16 (min normal 2^-126) immediately.
// ---------------------------------------------------------------------------

typedef unsigned int u32;
typedef __attribute__((ext_vector_type(16))) __bf16 v16bf;
typedef __attribute__((ext_vector_type(8)))  float  v8f;
typedef __attribute__((ext_vector_type(4)))  u32    u32x4;
typedef __attribute__((ext_vector_type(2)))  u32    u32x2;

#define BATCH  4096
#define IN     1024
#define OUTF   1024
#define GRID_G 20
#define LDK    40   // padded LDS row pitch in halves (32 data + 8 pad, 80B rows)

// sqrt(log2(e)) / SIGMA, SIGMA = 4/19:  exp(-d^2) = exp2(-(d*sqrt(log2e))^2)
#define ACO 5.70533144173064f
#define L2E 1.44269504088896f

#if __has_builtin(__builtin_amdgcn_exp2f)
#define EXP2_RAW(x) __builtin_amdgcn_exp2f(x)   // bare v_exp_f32
#else
#define EXP2_RAW(x) exp2f(x)
#endif

union FragBF {
  v16bf bf;
  u32x4 q[2];
};

struct Regs {
  float4 xv[4];
  float  b0[2][4];
  float  b1[2][4];
};

__device__ __forceinline__ u32 pack_bf16_pair(float lo, float hi) {
  // dest = { bf16(hi) , bf16(lo) }  (truncating f32->bf16: upper 16 bits)
  return __builtin_amdgcn_perm(__builtin_bit_cast(u32, hi),
                               __builtin_bit_cast(u32, lo), 0x07060302u);
}

// ---- global loads for one 128x32 A-tile (x) and 32x128 B-tile (weights) ----
__device__ __forceinline__
Regs load_tile(const float* __restrict__ xp,   // x + row0*IN + i0
               const float* __restrict__ bp,   // weight row k=0 (+col0)
               size_t bstride, int tid)
{
  Regs r;
  const int ar  = tid >> 3;        // 0..31 : x row group
  const int ac  = (tid & 7) * 4;   // 0..28 : k offset within chunk
  const int bn  = tid & 31;        // 0..31 : weight column lane
  const int bkp = tid >> 5;        // 0..7  : k-pair group
#pragma unroll
  for (int rr = 0; rr < 4; ++rr)
    r.xv[rr] = *(const float4*)(xp + (size_t)(ar + rr * 32) * IN + ac);
#pragma unroll
  for (int kk = 0; kk < 2; ++kk) {
    const float* r0 = bp + (size_t)(bkp + kk * 8) * 2 * bstride;
    const float* r1 = r0 + bstride;
#pragma unroll
    for (int nn = 0; nn < 4; ++nn) {
      r.b0[kk][nn] = r0[bn + nn * 32];   // coalesced: lanes sweep n
      r.b1[kk][nn] = r1[bn + nn * 32];
    }
  }
  return r;
}

// ---- basis/silu + fp32->bf16 convert + LDS stores ----
template<bool SPLINE>
__device__ __forceinline__
void convert_store(const Regs& r, float mga,
                   unsigned short* sA, unsigned short* sB, int tid)
{
  const int ar  = tid >> 3;
  const int ac  = (tid & 7) * 4;
  const int bn  = tid & 31;
  const int bkp = tid >> 5;

  // A tile, row-major [m][k], pitch LDK
#pragma unroll
  for (int rr = 0; rr < 4; ++rr) {
    const float f[4] = {r.xv[rr].x, r.xv[rr].y, r.xv[rr].z, r.xv[rr].w};
    float o[4];
#pragma unroll
    for (int e = 0; e < 4; ++e) {
      if (SPLINE) {
        const float v = __builtin_fmaf(f[e], ACO, -mga);
        o[e] = EXP2_RAW(-(v * v));                    // fma + mul(neg) + exp
      } else {
        const float ex = EXP2_RAW(-f[e] * L2E);       // silu = x/(1+e^-x)
        o[e] = f[e] * __builtin_amdgcn_rcpf(1.0f + ex);
      }
    }
    const u32x2 pk = { pack_bf16_pair(o[0], o[1]), pack_bf16_pair(o[2], o[3]) };
    *(u32x2*)&sA[(size_t)(ar + rr * 32) * LDK + ac] = pk;
  }

  // B tile, K-major per output column: sB[n][k]
#pragma unroll
  for (int kk = 0; kk < 2; ++kk) {
    const int k = (bkp + kk * 8) * 2;
#pragma unroll
    for (int nn = 0; nn < 4; ++nn) {
      const int n = bn + nn * 32;
      *(u32*)&sB[(size_t)n * LDK + k] = pack_bf16_pair(r.b0[kk][nn], r.b1[kk][nn]);
    }
  }
}

// ---- fragment loads (ISA 7.12.2 layouts) + 8 WMMAs per wave ----
__device__ __forceinline__
void compute(const unsigned short* sA, const unsigned short* sB,
             int lane, int wm, int wn, v8f acc[4][2])
{
  const int m16 = lane & 15;
  const int kh  = lane >> 4;

  FragBF a[4], b[2];
#pragma unroll
  for (int mt = 0; mt < 4; ++mt) {
    // A 16x32: lanes 0-15 -> K 0-7 & 16-23 ; lanes 16-31 -> K 8-15 & 24-31
    const int off = (wm * 64 + mt * 16 + m16) * LDK + kh * 8;
    a[mt].q[0] = *(const u32x4*)&sA[off];
    a[mt].q[1] = *(const u32x4*)&sA[off + 16];
  }
#pragma unroll
  for (int nt = 0; nt < 2; ++nt) {
    // B 32x16: lanes 0-15 -> K 0-7 & 16-23 of col n ; lanes 16-31 -> K 8-15 & 24-31
    const int off = (wn * 32 + nt * 16 + m16) * LDK + kh * 8;
    b[nt].q[0] = *(const u32x4*)&sB[off];
    b[nt].q[1] = *(const u32x4*)&sB[off + 16];
  }
#pragma unroll
  for (int mt = 0; mt < 4; ++mt)
#pragma unroll
    for (int nt = 0; nt < 2; ++nt)
      acc[mt][nt] = __builtin_amdgcn_wmma_f32_16x16x32_bf16(
          false, a[mt].bf, false, b[nt].bf, (short)0, acc[mt][nt], false, false);
}

// ---- one 32-step K phase (fixed g, or the base path), software pipelined ----
template<bool SPLINE>
__device__ __forceinline__
void run_phase(const float* __restrict__ xbase,   // x + row0*IN
               const float* __restrict__ bbase,   // weight k=0 row (+col0)
               size_t bstride, float mga,
               unsigned short* sA0, unsigned short* sB0,
               unsigned short* sA1, unsigned short* sB1,
               int tid, int lane, int wm, int wn, v8f acc[4][2])
{
  Regs r = load_tile(xbase, bbase, bstride, tid);
  convert_store<SPLINE>(r, mga, sA0, sB0, tid);
  __syncthreads();

#pragma unroll 1
  for (int ic = 0; ic < 31; ++ic) {
    unsigned short* cA = (ic & 1) ? sA1 : sA0;
    unsigned short* cB = (ic & 1) ? sB1 : sB0;
    unsigned short* nA = (ic & 1) ? sA0 : sA1;
    unsigned short* nB = (ic & 1) ? sB0 : sB1;

    r = load_tile(xbase + (size_t)(ic + 1) * 32,
                  bbase + (size_t)(ic + 1) * 32 * bstride, bstride, tid);
    compute(cA, cB, lane, wm, wn, acc);         // WMMAs on current buffer ...
    convert_store<SPLINE>(r, mga, nA, nB, tid); // ... overlap next tile's VALU
    __syncthreads();                            // single barrier per K-step
  }
  // last step of the phase (buffer 1); no trailing barrier needed: the next
  // phase's prologue writes buffer 0, which all waves finished reading
  // before the last barrier above.
  compute(sA1, sB1, lane, wm, wn, acc);
}

__global__ __launch_bounds__(256)
void kan_wmma_kernel(const float* __restrict__ x,
                     const float* __restrict__ base_w,   // [IN][OUT]
                     const float* __restrict__ spline_w, // [IN][G][OUT]
                     const float* __restrict__ mu,       // [G]
                     float* __restrict__ out)            // [BATCH][OUT]
{
  __shared__ __align__(16) unsigned short sA0[128 * LDK];  // 4 x 10 KB
  __shared__ __align__(16) unsigned short sB0[128 * LDK];
  __shared__ __align__(16) unsigned short sA1[128 * LDK];
  __shared__ __align__(16) unsigned short sB1[128 * LDK];

  const int tid  = threadIdx.x;
  const int lane = tid & 31;
  const int wave = tid >> 5;
  const int wm   = wave & 1;    // 2 waves along M
  const int wn   = wave >> 1;   // 4 waves along N

  const size_t row0 = (size_t)blockIdx.y * 128;
  const size_t col0 = (size_t)blockIdx.x * 128;

  v8f acc[4][2] = {};

  // ---- spline part: K = 20480, ordered k = g*IN + i ----
#pragma unroll 1
  for (int g = 0; g < GRID_G; ++g) {
    const float mga = mu[g] * ACO;
    run_phase<true>(x + row0 * IN,
                    spline_w + (size_t)g * OUTF + col0,   // + i*G*OUT per k
                    (size_t)(GRID_G * OUTF), mga,
                    sA0, sB0, sA1, sB1, tid, lane, wm, wn, acc);
  }

  // ---- base part: K = 1024, A = silu(x), B = base_weight ----
  run_phase<false>(x + row0 * IN,
                   base_w + col0,
                   (size_t)OUTF, 0.0f,
                   sA0, sB0, sA1, sB1, tid, lane, wm, wn, acc);

  // ---- epilogue: C/D layout = VGPR j -> M = 8*(lane/16)+j, N = lane%16 ----
  const int m16 = lane & 15;
  const int kh  = lane >> 4;
  float* op = out + (row0 + (size_t)wm * 64) * OUTF + col0 + wn * 32;
#pragma unroll
  for (int mt = 0; mt < 4; ++mt)
#pragma unroll
    for (int nt = 0; nt < 2; ++nt)
#pragma unroll
      for (int j = 0; j < 8; ++j)
        op[(size_t)(mt * 16 + kh * 8 + j) * OUTF + nt * 16 + m16] = acc[mt][nt][j];
}

extern "C" void kernel_launch(void* const* d_in, const int* in_sizes, int n_in,
                              void* d_out, int out_size, void* d_ws, size_t ws_size,
                              hipStream_t stream) {
  (void)in_sizes; (void)n_in; (void)d_ws; (void)ws_size; (void)out_size;
  const float* x  = (const float*)d_in[0];
  const float* bw = (const float*)d_in[1];
  const float* sw = (const float*)d_in[2];
  const float* mu = (const float*)d_in[3];
  float* out = (float*)d_out;

  dim3 grid(OUTF / 128, BATCH / 128);   // 8 x 32 = 256 workgroups
  kan_wmma_kernel<<<grid, dim3(256), 0, stream>>>(x, bw, sw, mu, out);
}
